// SlotAttention_66632122630478
// MI455X (gfx1250) — compile-verified, matrix operand order checked by reading, not run on hardware
//
#include <hip/hip_runtime.h>
#include <stdint.h>

// ------------------------- problem constants -------------------------
#define B_     64
#define K_     4096
#define D_     256
#define Q_     8
#define H_     4
#define DH_    64
#define HM_    512
#define G3_    768
#define NITER  3
#define ROWS_  (B_*Q_)   // 512
#define BH_    (B_*H_)   // 256
#define EPS_   1e-8f
#define LNEPS_ 1e-5f

typedef __attribute__((ext_vector_type(16))) __bf16 v16bf;
typedef __attribute__((ext_vector_type(8)))  __bf16 v8bf;
typedef __attribute__((ext_vector_type(8)))  float  v8f;
typedef unsigned short ushort_t;

// ------------------------- helpers -------------------------
static __device__ __forceinline__ unsigned short f2u16(float f) {
  unsigned u = __builtin_bit_cast(unsigned, f);
  u = (u + 0x7FFFu + ((u >> 16) & 1u)) >> 16;   // RNE to bf16
  return (unsigned short)u;
}
static __device__ __forceinline__ __bf16 f2bf(float f) {
  unsigned short s = f2u16(f);
  return __builtin_bit_cast(__bf16, s);
}
static __device__ __forceinline__ v8bf ld8(const ushort_t* p) {
  uint4 u = *(const uint4*)p;
  return __builtin_bit_cast(v8bf, u);
}
static __device__ __forceinline__ v16bf mk16(v8bf lo, v8bf hi) {
  v16bf r;
#pragma unroll
  for (int i = 0; i < 8; i++) { r[i] = lo[i]; r[i + 8] = hi[i]; }
  return r;
}
static __device__ __forceinline__ v16bf zero16() {
  v16bf r;
#pragma unroll
  for (int i = 0; i < 16; i++) r[i] = f2bf(0.f);
  return r;
}
static __device__ __forceinline__ v8f zero8() {
  v8f r;
#pragma unroll
  for (int i = 0; i < 8; i++) r[i] = 0.f;
  return r;
}
#define WMMA_BF16(a, b, c) \
  __builtin_amdgcn_wmma_f32_16x16x32_bf16(false, (a), false, (b), (short)0, (c), false, false)

static __device__ __forceinline__ float wave_sum(float x) {
#pragma unroll
  for (int off = 16; off > 0; off >>= 1) x += __shfl_xor(x, off, 32);
  return x;
}

// ------------------------- kernel 0: weight prep -------------------------
// Wt (bf16, [512][256]) = transpose(Wk)*scale || transpose(Wv)
__global__ __launch_bounds__(256) void prep_wkv_kernel(
    const float* __restrict__ Wk, const float* __restrict__ Wv, ushort_t* __restrict__ Wt) {
  int idx = blockIdx.x * 256 + threadIdx.x;   // 512*256
  int n = idx >> 8;
  int d = idx & 255;
  float v = (n < D_) ? Wk[d * D_ + n] * 0.125f   // dh^-0.5 folded into k
                     : Wv[d * D_ + (n - D_)];
  Wt[(size_t)n * D_ + d] = f2u16(v);
}

// ------------------------- kernel 1: LN(inputs) + K/V projection -------------------------
// grid = B*K/128; block = 256 (8 waves). Produces kbf [B,K,D] bf16 and vT [B,H,dh,K] bf16.
// Ping-pong B-fragment buffers (unroll x2): 16 global_load_b128 overlap 8 WMMAs, no reg copies.
#define TM_ 128
__global__ __launch_bounds__(256) void ln_kv_kernel(
    const float* __restrict__ inp, const float* __restrict__ lnw, const float* __restrict__ lnb,
    const ushort_t* __restrict__ Wt, ushort_t* __restrict__ kbf, ushort_t* __restrict__ vT) {
  __shared__ __align__(16) ushort_t xs[TM_][264];   // 264 pitch: conflict-free b128 frag loads
  const int tid  = threadIdx.x;
  const int wave = tid >> 5, lane = tid & 31;
  const int lr = lane & 15, hi = lane >> 4;
  const int block_row0 = blockIdx.x * TM_;

  // ---- LayerNorm 128 rows -> bf16 tile in LDS ----
  for (int r = wave; r < TM_; r += 8) {
    const float* row = inp + (size_t)(block_row0 + r) * D_;
    float v[8]; float s = 0.f;
#pragma unroll
    for (int i = 0; i < 8; i++) { v[i] = row[lane + 32 * i]; s += v[i]; }
    s = wave_sum(s);
    float mu = s * (1.f / D_);
    float vs = 0.f;
#pragma unroll
    for (int i = 0; i < 8; i++) { float d = v[i] - mu; vs += d * d; }
    vs = wave_sum(vs);
    float rstd = rsqrtf(vs * (1.f / D_) + LNEPS_);
#pragma unroll
    for (int i = 0; i < 8; i++) {
      int c = lane + 32 * i;
      xs[r][c] = f2u16((v[i] - mu) * rstd * lnw[c] + lnb[c]);
    }
  }
  __syncthreads();

  // ---- WMMA GEMM: wave owns M-tile = wave; loops 32 N-tiles over [Wk*s | Wv] ----
  v16bf a[8];
#pragma unroll
  for (int kc = 0; kc < 8; kc++) {
    const ushort_t* p = &xs[wave * 16 + lr][kc * 32 + hi * 8];
    a[kc] = mk16(ld8(p), ld8(p + 16));
  }
  const int bb = block_row0 / K_;                      // batch (TM_ divides K_)
  const int kpos0 = (block_row0 % K_) + wave * 16 + hi * 8;

  auto loadB = [&](v16bf* dst, int nt) {
    const ushort_t* bp = Wt + (size_t)(nt * 16 + lr) * D_ + hi * 8;
    __builtin_prefetch(bp + 2 * D_, 0, 1);
#pragma unroll
    for (int kc = 0; kc < 8; kc++)
      dst[kc] = mk16(ld8(bp + kc * 32), ld8(bp + kc * 32 + 16));
  };
  auto tile = [&](int nt, const v16bf* bfr) {
    v8f c = zero8();
#pragma unroll
    for (int kc = 0; kc < 8; kc++) c = WMMA_BF16(a[kc], bfr[kc], c);
    const int col = nt * 16 + lr;
    if (col < D_) {
      // k: row-major [B*K, D] bf16 (rows = kv positions)
      const int rbase = block_row0 + wave * 16 + hi * 8;
#pragma unroll
      for (int i = 0; i < 8; i++)
        kbf[(size_t)(rbase + i) * D_ + col] = f2u16(c[i]);
    } else {
      // v transposed: vT[(b*D + dcol)*K + kpos], lane's 8 accum values = contiguous kpos
      const int dcol = col - D_;
      __align__(16) ushort_t tmp[8];
#pragma unroll
      for (int i = 0; i < 8; i++) tmp[i] = f2u16(c[i]);
      size_t out = ((size_t)bb * D_ + dcol) * K_ + kpos0;
      *(uint4*)(vT + out) = *(const uint4*)tmp;
    }
  };

  v16bf b0[8], b1[8];
  loadB(b0, 0);
  for (int nt = 0; nt < 32; nt += 2) {
    loadB(b1, nt + 1);                    // overlaps tile(nt)
    tile(nt, b0);
    if (nt + 2 < 32) loadB(b0, nt + 2);   // overlaps tile(nt+1)
    tile(nt + 1, b1);
  }
}

// ------------------------- kernel: LayerNorm rows (slots / mlp) -------------------------
__global__ __launch_bounds__(256) void ln_rows_kernel(
    const float* __restrict__ src, const float* __restrict__ w, const float* __restrict__ b,
    float* __restrict__ dst) {
  const int wave = threadIdx.x >> 5, lane = threadIdx.x & 31;
  const int row = blockIdx.x * 8 + wave;
  const float* s = src + (size_t)row * D_;
  float v[8]; float sum = 0.f;
#pragma unroll
  for (int i = 0; i < 8; i++) { v[i] = s[lane + 32 * i]; sum += v[i]; }
  sum = wave_sum(sum);
  float mu = sum * (1.f / D_);
  float vs = 0.f;
#pragma unroll
  for (int i = 0; i < 8; i++) { float d = v[i] - mu; vs += d * d; }
  vs = wave_sum(vs);
  float rstd = rsqrtf(vs * (1.f / D_) + LNEPS_);
  float* o = dst + (size_t)row * D_;
#pragma unroll
  for (int i = 0; i < 8; i++) {
    int c = lane + 32 * i;
    o[c] = (v[i] - mu) * rstd * w[c] + b[c];
  }
}

// ------------------------- kernel: small fp32 GEMM (q / GRU / MLP) -------------------------
// C[row][c] = act( sum_d A[row][d]*W + bias[c] ) (+ Cadd); wlayout 0: W[d*ldw+c], 1: W[c*ldw+d]
__global__ __launch_bounds__(256) void small_gemm_kernel(
    const float* __restrict__ A, int kd,
    const float* __restrict__ W, int wlayout, int ldw,
    const float* __restrict__ bias, const float* __restrict__ Cadd,
    float* __restrict__ Cout, ushort_t* __restrict__ Cbf, int cols, int act) {
  __shared__ float srow[512];
  const int row = blockIdx.x, tid = threadIdx.x;
  for (int i = tid; i < kd; i += 256) srow[i] = A[(size_t)row * kd + i];
  __syncthreads();
  for (int c = tid; c < cols; c += 256) {
    float acc = bias ? bias[c] : 0.f;
    if (wlayout == 0) {
      const float* wp = W + c;
#pragma unroll 4
      for (int d = 0; d < kd; d++) acc += srow[d] * wp[(size_t)d * ldw];
    } else {
      const float* wp = W + (size_t)c * ldw;
#pragma unroll 4
      for (int d = 0; d < kd; d++) acc += srow[d] * wp[d];
    }
    if (act == 1) acc = fmaxf(acc, 0.f);
    if (Cadd) acc += Cadd[(size_t)row * cols + c];
    if (Cout) Cout[(size_t)row * cols + c] = acc;
    if (Cbf)  Cbf[(size_t)row * cols + c] = f2u16(acc);
  }
}

// ------------------------- kernel: logits = k . q (WMMA bf16) -------------------------
// grid (8, B*H); block 8 waves; writes attnT [B,H,Q,K] fp32. Ping-pong A-fragments over tiles.
__global__ __launch_bounds__(256) void logits_kernel(
    const ushort_t* __restrict__ kbf, const ushort_t* __restrict__ qbf,
    float* __restrict__ attnT) {
  const int bh = blockIdx.y, b = bh / H_, h = bh % H_;
  const int tid = threadIdx.x, wave = tid >> 5, lane = tid & 31;
  const int lr = lane & 15, hi = lane >> 4;

  // B fragment: columns = slots q (zero-pad q>=8), contraction = dh
  v16bf bq[2];
  const ushort_t* qp = qbf + ((size_t)(b * Q_ + lr)) * D_ + h * DH_ + hi * 8;
#pragma unroll
  for (int kc = 0; kc < 2; kc++)
    bq[kc] = (lr < Q_) ? mk16(ld8(qp + kc * 32), ld8(qp + kc * 32 + 16)) : zero16();

  auto loadA = [&](v16bf& a0, v16bf& a1, int t) {
    const int ktile = blockIdx.x * 32 + wave * 4 + t;
    const ushort_t* ap = kbf + ((size_t)(b * K_ + ktile * 16 + lr)) * D_ + h * DH_ + hi * 8;
    a0 = mk16(ld8(ap), ld8(ap + 16));
    a1 = mk16(ld8(ap + 32), ld8(ap + 48));
  };
  auto tile = [&](int t, v16bf a0, v16bf a1) {
    v8f c = zero8();
    c = WMMA_BF16(a0, bq[0], c);
    c = WMMA_BF16(a1, bq[1], c);
    if (lr < Q_) {
      const int ktile = blockIdx.x * 32 + wave * 4 + t;
      size_t out = ((size_t)(bh * Q_ + lr)) * K_ + ktile * 16 + hi * 8;
#pragma unroll
      for (int i = 0; i < 8; i++) attnT[out + i] = c[i];
    }
  };

  v16bf p0, p1, q0, q1;
  loadA(p0, p1, 0);
  for (int t = 0; t < 4; t += 2) {
    loadA(q0, q1, t + 1);                  // overlaps tile(t)
    tile(t, p0, p1);
    if (t + 2 < 4) loadA(p0, p1, t + 2);   // overlaps tile(t+1)
    tile(t + 1, q0, q1);
  }
}

// ------------------------- kernel: joint softmax over (H,Q) + partial K-sums -------------------------
// grid (16, B); thread owns one (b, kpos). Deterministic block partials (no atomics).
__global__ __launch_bounds__(256) void softmax_kernel(
    float* __restrict__ attnT, float* __restrict__ partial) {
  __shared__ float red[32][8];
  const int tid = threadIdx.x, wave = tid >> 5, lane = tid & 31;
  const int b = blockIdx.y;
  const int kpos = blockIdx.x * 256 + tid;
  const size_t base = (size_t)b * 32 * K_ + kpos;

  float v[32]; float m = -3.4e38f;
#pragma unroll
  for (int j = 0; j < 32; j++) { v[j] = attnT[base + (size_t)j * K_]; m = fmaxf(m, v[j]); }
  float s = 0.f;
#pragma unroll
  for (int j = 0; j < 32; j++) { v[j] = expf(v[j] - m); s += v[j]; }
  float inv = 1.f / s;
#pragma unroll
  for (int j = 0; j < 32; j++) {
    float a = v[j] * inv + EPS_;
    attnT[base + (size_t)j * K_] = a;
    float ws = wave_sum(a);
    if (lane == 0) red[j][wave] = ws;
  }
  __syncthreads();
  if (tid < 32) {
    float t = 0.f;
#pragma unroll
    for (int w = 0; w < 8; w++) t += red[tid][w];
    partial[((size_t)b * 16 + blockIdx.x) * 32 + tid] = t;
  }
}

__global__ __launch_bounds__(256) void denom_reduce_kernel(
    const float* __restrict__ partial, float* __restrict__ denom) {
  int idx = blockIdx.x * 256 + threadIdx.x;   // 2048 = B*H*Q
  if (idx >= B_ * 32) return;
  int b = idx >> 5, j = idx & 31;
  float s = 0.f;
#pragma unroll
  for (int blk = 0; blk < 16; blk++) s += partial[((size_t)b * 16 + blk) * 32 + j];
  denom[idx] = s;
}

// ------------------------- kernel: updates = attn^T . v (WMMA bf16) -------------------------
// grid = B*H; block 8 waves split the K loop (ping-pong pipelined); LDS reduce; fold 1/denom.
__global__ __launch_bounds__(256) void updates_kernel(
    const float* __restrict__ attnT, const ushort_t* __restrict__ vT,
    const float* __restrict__ denom, float* __restrict__ updates) {
  __shared__ float red[8][4][16][16];
  const int bh = blockIdx.x, b = bh / H_, h = bh % H_;
  const int tid = threadIdx.x, wave = tid >> 5, lane = tid & 31;
  const int lr = lane & 15, hi = lane >> 4;

  auto loadA = [&](v16bf& dst, int kbase) {
    if (lr < Q_) {
      const float* ap = attnT + ((size_t)(bh * Q_ + lr)) * K_ + kbase + hi * 8;
#pragma unroll
      for (int i = 0; i < 8; i++) { dst[i] = f2bf(ap[i]); dst[8 + i] = f2bf(ap[16 + i]); }
    } else {
      dst = zero16();
    }
  };
  auto loadBv = [&](v16bf* dst, int kbase) {
#pragma unroll
    for (int t = 0; t < 4; t++) {
      const ushort_t* bp = vT + ((size_t)(bh * DH_) + t * 16 + lr) * K_ + kbase + hi * 8;
      dst[t] = mk16(ld8(bp), ld8(bp + 16));
    }
  };

  v8f acc[4] = { zero8(), zero8(), zero8(), zero8() };
  v16bf a0, a1;
  v16bf b0[4], b1[4];
  loadA(a0, wave * 32);
  loadBv(b0, wave * 32);
  for (int c = wave; c < K_ / 32; c += 16) {
    const int c1 = c + 8;                  // always < 128 (c <= wave+112)
    loadA(a1, c1 * 32);                    // overlaps WMMAs on (a0,b0)
    loadBv(b1, c1 * 32);
#pragma unroll
    for (int t = 0; t < 4; t++) acc[t] = WMMA_BF16(a0, b0[t], acc[t]);
    const int c2 = c + 16;
    if (c2 < K_ / 32) { loadA(a0, c2 * 32); loadBv(b0, c2 * 32); }  // overlaps (a1,b1)
#pragma unroll
    for (int t = 0; t < 4; t++) acc[t] = WMMA_BF16(a1, b1[t], acc[t]);
  }
#pragma unroll
  for (int t = 0; t < 4; t++)
#pragma unroll
    for (int i = 0; i < 8; i++) red[wave][t][hi * 8 + i][lr] = acc[t][i];
  __syncthreads();
  for (int o = tid; o < Q_ * DH_; o += 256) {
    int q = o >> 6, d = o & 63, t = d >> 4, n = d & 15;
    float s = 0.f;
#pragma unroll
    for (int w = 0; w < 8; w++) s += red[w][t][q][n];
    updates[((size_t)b * Q_ + q) * D_ + h * DH_ + d] = s / denom[bh * Q_ + q];
  }
}

// ------------------------- kernel: GRU gate combine -------------------------
__global__ __launch_bounds__(256) void gru_gate_kernel(
    const float* __restrict__ gi, const float* __restrict__ gh, float* __restrict__ slots) {
  int idx = blockIdx.x * 256 + threadIdx.x;    // 512*256
  int r = idx >> 8, d = idx & 255;
  const float* gir = gi + (size_t)r * G3_;
  const float* ghr = gh + (size_t)r * G3_;
  float rr = 1.f / (1.f + expf(-(gir[d] + ghr[d])));
  float z  = 1.f / (1.f + expf(-(gir[D_ + d] + ghr[D_ + d])));
  float n  = tanhf(gir[2 * D_ + d] + rr * ghr[2 * D_ + d]);
  float hp = slots[idx];
  slots[idx] = (1.f - z) * n + z * hp;
}

// ------------------------- kernel: final attn output (mean over heads, renorm) -------------------------
__global__ __launch_bounds__(256) void attn_out_kernel(
    const float* __restrict__ attnT, const float* __restrict__ denom, float* __restrict__ out) {
  int idx = blockIdx.x * 256 + threadIdx.x;    // B*Q*K
  int kpos = idx & (K_ - 1);
  int bq = idx >> 12;                          // K_ = 4096
  int b = bq >> 3, q = bq & 7;
  float s = 0.f;
#pragma unroll
  for (int h = 0; h < H_; h++) {
    int j = b * 32 + h * 8 + q;
    s += attnT[(size_t)j * K_ + kpos] / denom[j];
  }
  out[idx] = s * (1.f / H_);
}

// ------------------------- host orchestration -------------------------
extern "C" void kernel_launch(void* const* d_in, const int* in_sizes, int n_in,
                              void* d_out, int out_size, void* d_ws, size_t ws_size,
                              hipStream_t stream) {
  const float* inputs     = (const float*)d_in[0];
  const float* slots_init = (const float*)d_in[1];
  const float* Wq         = (const float*)d_in[2];
  const float* Wk         = (const float*)d_in[3];
  const float* Wv         = (const float*)d_in[4];
  const float* ln_in_w    = (const float*)d_in[5];
  const float* ln_in_b    = (const float*)d_in[6];
  const float* ln_sl_w    = (const float*)d_in[7];
  const float* ln_sl_b    = (const float*)d_in[8];
  const float* ln_mlp_w   = (const float*)d_in[9];
  const float* ln_mlp_b   = (const float*)d_in[10];
  const float* gru_Wih    = (const float*)d_in[11];
  const float* gru_Whh    = (const float*)d_in[12];
  const float* gru_bih    = (const float*)d_in[13];
  const float* gru_bhh    = (const float*)d_in[14];
  const float* mlp_W1     = (const float*)d_in[15];
  const float* mlp_b1     = (const float*)d_in[16];
  const float* mlp_W2     = (const float*)d_in[17];
  const float* mlp_b2     = (const float*)d_in[18];

  char* ws = (char*)d_ws;
  const size_t OFF_KBF   = 0;                                  // 134217728
  const size_t OFF_VT    = OFF_KBF  + (size_t)B_ * K_ * D_ * 2;
  const size_t OFF_ATTN  = OFF_VT   + (size_t)B_ * K_ * D_ * 2;
  const size_t OFF_WT    = OFF_ATTN + (size_t)BH_ * Q_ * K_ * 4;
  const size_t OFF_QBF   = OFF_WT   + (size_t)2 * D_ * D_ * 2;
  const size_t OFF_SLOTS = OFF_QBF  + (size_t)ROWS_ * D_ * 2;
  const size_t OFF_SN    = OFF_SLOTS+ (size_t)ROWS_ * D_ * 4;
  const size_t OFF_UPD   = OFF_SN   + (size_t)ROWS_ * D_ * 4;
  const size_t OFF_GI    = OFF_UPD  + (size_t)ROWS_ * D_ * 4;
  const size_t OFF_GH    = OFF_GI   + (size_t)ROWS_ * G3_ * 4;
  const size_t OFF_HID   = OFF_GH   + (size_t)ROWS_ * G3_ * 4;
  const size_t OFF_M     = OFF_HID  + (size_t)ROWS_ * HM_ * 4;
  const size_t OFF_DEN   = OFF_M    + (size_t)ROWS_ * D_ * 4;
  const size_t OFF_PART  = OFF_DEN  + (size_t)B_ * 32 * 4;

  ushort_t* kbf   = (ushort_t*)(ws + OFF_KBF);
  ushort_t* vT    = (ushort_t*)(ws + OFF_VT);
  float*    attnT = (float*)(ws + OFF_ATTN);
  ushort_t* Wt    = (ushort_t*)(ws + OFF_WT);
  ushort_t* qbf   = (ushort_t*)(ws + OFF_QBF);
  float*    slots = (float*)(ws + OFF_SLOTS);
  float*    sn    = (float*)(ws + OFF_SN);
  float*    upd   = (float*)(ws + OFF_UPD);
  float*    gi    = (float*)(ws + OFF_GI);
  float*    gh    = (float*)(ws + OFF_GH);
  float*    hid   = (float*)(ws + OFF_HID);
  float*    mbuf  = (float*)(ws + OFF_M);
  float*    den   = (float*)(ws + OFF_DEN);
  float*    part  = (float*)(ws + OFF_PART);

  // ---- once: weight prep + LN + K/V projection ----
  prep_wkv_kernel<<<512, 256, 0, stream>>>(Wk, Wv, Wt);
  ln_kv_kernel<<<(B_ * K_) / TM_, 256, 0, stream>>>(inputs, ln_in_w, ln_in_b, Wt, kbf, vT);
  hipMemcpyAsync(slots, slots_init, (size_t)ROWS_ * D_ * 4, hipMemcpyDeviceToDevice, stream);

  for (int it = 0; it < NITER; ++it) {
    // slots LN + q projection (q in bf16, ready for WMMA B-fragments)
    ln_rows_kernel<<<ROWS_ / 8, 256, 0, stream>>>(slots, ln_sl_w, ln_sl_b, sn);
    small_gemm_kernel<<<ROWS_, 256, 0, stream>>>(sn, D_, Wq, 0, D_, nullptr, nullptr,
                                                 nullptr, qbf, D_, 0);
    // attention
    logits_kernel<<<dim3(8, BH_), 256, 0, stream>>>(kbf, qbf, attnT);
    softmax_kernel<<<dim3(16, B_), 256, 0, stream>>>(attnT, part);
    denom_reduce_kernel<<<8, 256, 0, stream>>>(part, den);
    updates_kernel<<<BH_, 256, 0, stream>>>(attnT, vT, den, upd);
    // GRU
    small_gemm_kernel<<<ROWS_, 256, 0, stream>>>(upd, D_, gru_Wih, 1, D_, gru_bih, nullptr,
                                                 gi, nullptr, G3_, 0);
    small_gemm_kernel<<<ROWS_, 256, 0, stream>>>(slots, D_, gru_Whh, 1, D_, gru_bhh, nullptr,
                                                 gh, nullptr, G3_, 0);
    gru_gate_kernel<<<(ROWS_ * D_) / 256, 256, 0, stream>>>(gi, gh, slots);
    // MLP residual
    ln_rows_kernel<<<ROWS_ / 8, 256, 0, stream>>>(slots, ln_mlp_w, ln_mlp_b, mbuf);
    small_gemm_kernel<<<ROWS_, 256, 0, stream>>>(mbuf, D_, mlp_W1, 0, HM_, mlp_b1, nullptr,
                                                 hid, nullptr, HM_, 1);
    small_gemm_kernel<<<ROWS_, 256, 0, stream>>>(hid, HM_, mlp_W2, 0, D_, mlp_b2, slots,
                                                 slots, nullptr, D_, 0);
  }

  // ---- outputs: slots [B,Q,D] then attn-mean [B,Q,K] ----
  float* out_f = (float*)d_out;
  hipMemcpyAsync(out_f, slots, (size_t)ROWS_ * D_ * 4, hipMemcpyDeviceToDevice, stream);
  attn_out_kernel<<<(B_ * Q_ * K_) / 256, 256, 0, stream>>>(attnT, den, out_f + ROWS_ * D_);
}
